// MultiHeadedAttention_46291157517079
// MI455X (gfx1250) — compile-verified, matrix operand order checked by reading, not run on hardware
//
#include <hip/hip_runtime.h>
#include <hip/hip_bf16.h>

// ---------------------------------------------------------------------------
// MHA forward for B=2, T=2048, D=1024, H=16, DK=64 on gfx1250 (MI455X).
// All GEMMs run through v_wmma_f32_16x16x32_f16 (f16 in, f32 accumulate).
// GEMM tiles are staged global->LDS with GLOBAL_LOAD_ASYNC_TO_LDS_B128
// (ASYNCcnt) in a double-buffered pipeline.
// ---------------------------------------------------------------------------

typedef __attribute__((ext_vector_type(16))) _Float16 v16h;
typedef __attribute__((ext_vector_type(8)))  _Float16 v8h;
typedef __attribute__((ext_vector_type(4)))  _Float16 v4h;
typedef __attribute__((ext_vector_type(8)))  float    v8f;

#define B_  2
#define T_  2048
#define H_  16
#define DK_ 64
#define D_  1024
#define M_  (B_ * T_)   // 4096 rows for the projection GEMMs

__device__ __forceinline__ v8f wmma16(v16h a, v16h b, v8f c) {
  // (neg_a, A, neg_b, B, c_mod, C, reuse_a, reuse_b)
  return __builtin_amdgcn_wmma_f32_16x16x32_f16(false, a, false, b, (short)0, c,
                                                false, false);
}

// Async DMA of one 16-byte chunk: global -> LDS, tracked with ASYNCcnt.
// VDST = per-lane LDS byte offset (low 32 bits of the generic LDS pointer,
// per the CDNA5 aperture mapping), VADDR = 64-bit global address.
__device__ __forceinline__ void async_copy_b128(void* lds_ptr,
                                                const void* gptr) {
  const unsigned loff = (unsigned)(unsigned long long)lds_ptr;
  asm volatile("global_load_async_to_lds_b128 %0, %1, off" ::"v"(loff),
               "v"(gptr)
               : "memory");
}
__device__ __forceinline__ void wait_async0() {
  asm volatile("s_wait_asynccnt 0x0" ::: "memory");
}

// Load a 16x32 f16 WMMA A/B fragment from LDS stored [row][k], k contiguous,
// row stride STRIDE halves (STRIDE % 8 == 0 so each half-row chunk is a
// 16-byte-aligned ds_load_b128).
// Lane L (L<16): row = row0+L, K = {kh..kh+7, 16+kh..16+kh+7} with kh=0.
// Lane L+16:    same row, kh=8.  Matches CDNA5 16-bit A layout (ISA 7.12.2).
template <int STRIDE>
__device__ __forceinline__ v16h load_frag_lds(const _Float16* base, int row0,
                                              int kOff) {
  const int lane = threadIdx.x & 31;
  const _Float16* p =
      base + (size_t)(row0 + (lane & 15)) * STRIDE + kOff + ((lane >> 4) << 3);
  v8h lo = *(const v8h*)(p);
  v8h hi = *(const v8h*)(p + 16);
  return __builtin_shufflevector(lo, hi, 0, 1, 2, 3, 4, 5, 6, 7, 8, 9, 10, 11,
                                 12, 13, 14, 15);
}

// Same fragment shape, loaded directly from global memory with row stride 64
// halves (the [*, 64] head layout of Q/K: row = head-row, k-dim = feature).
__device__ __forceinline__ v16h load_frag_g64(const _Float16* base, int kOff) {
  const int lane = threadIdx.x & 31;
  const _Float16* p =
      base + (size_t)(lane & 15) * DK_ + kOff + ((lane >> 4) << 3);
  v8h lo = *(const v8h*)(p);
  v8h hi = *(const v8h*)(p + 16);
  return __builtin_shufflevector(lo, hi, 0, 1, 2, 3, 4, 5, 6, 7, 8, 9, 10, 11,
                                 12, 13, 14, 15);
}

// ---------------------------------------------------------------------------
// Kernel 0a: tanh(learnable_mask) -> f16, computed once, reused by all (b,h).
// ---------------------------------------------------------------------------
__global__ void tanh_mask_kernel(const float* __restrict__ mask,
                                 _Float16* __restrict__ out, int n) {
  int i = blockIdx.x * blockDim.x + threadIdx.x;
  if (i < n) out[i] = (_Float16)tanhf(mask[i]);
}

// ---------------------------------------------------------------------------
// Kernel 0b: f32 -> f16 elementwise (activations), 4 elems/thread.
// ---------------------------------------------------------------------------
__global__ void cvt_f16_kernel(const float* __restrict__ in,
                               _Float16* __restrict__ out, int n4) {
  int i = blockIdx.x * blockDim.x + threadIdx.x;
  if (i < n4) {
    float4 v = ((const float4*)in)[i];
    v4h h;
    h.x = (_Float16)v.x; h.y = (_Float16)v.y;
    h.z = (_Float16)v.z; h.w = (_Float16)v.w;
    ((v4h*)out)[i] = h;
  }
}

// ---------------------------------------------------------------------------
// Kernel 0c: W[k][n] f32 -> Wt[n][k] f16 (tiled 32x32 transpose via LDS) so
// the GEMM B-operand is k-contiguous and async-DMA-friendly.
// ---------------------------------------------------------------------------
__global__ __launch_bounds__(256) void transpose_cvt_kernel(
    const float* __restrict__ W, _Float16* __restrict__ Wt) {
  __shared__ float tile[32][33];
  const int tx = threadIdx.x, ty = threadIdx.y;
  const int n0 = blockIdx.x * 32, k0 = blockIdx.y * 32;
  for (int i = ty; i < 32; i += 8)
    tile[i][tx] = W[(size_t)(k0 + i) * D_ + n0 + tx];
  __syncthreads();
  for (int i = ty; i < 32; i += 8)
    Wt[(size_t)(n0 + i) * D_ + k0 + tx] = (_Float16)tile[tx][i];
}

// ---------------------------------------------------------------------------
// Kernel 1/4: C = A @ W + bias, with A [4096 x 1024] f16 row-major and
// Wt [1024(n) x 1024(k)] f16 (pre-transposed).  Tile 128x64, BK=32, 8 waves,
// each wave a 32x32 block = 2x2 WMMA tiles.  Tiles staged by
// global_load_async_to_lds_b128, double-buffered (issue k+1 while computing k).
// HEADS=true : scatter f16 into [B,H,T,64] head layout, scale folded in.
// HEADS=false: write f32 row-major [4096 x 1024] (final output).
// ---------------------------------------------------------------------------
#define BM 128
#define BN 64
#define BK 32
#define LS 40  // LDS row stride in halves (32 + 8 pad; 80B, 16B aligned)

template <bool HEADS>
__global__ __launch_bounds__(256) void gemm_wmma_async(
    const _Float16* __restrict__ A, const _Float16* __restrict__ Wt,
    const float* __restrict__ bias, void* __restrict__ outp, float scale) {
  __shared__ _Float16 sA[2][BM * LS];
  __shared__ _Float16 sB[2][BN * LS];

  const int tid = threadIdx.x;
  const int lane = tid & 31;
  const int wid = tid >> 5;
  const int wr = wid & 3;   // wave row 0..3  -> 32-row stripe
  const int wc = wid >> 2;  // wave col 0..1  -> 32-col stripe
  const int m0 = blockIdx.x * BM;
  const int n0 = blockIdx.y * BN;

  // Issue async copies for the k0 tile into buffer `buf`.
  // A tile: 128 rows x 4 16B-chunks = 512 chunks (2/thread);
  // B tile:  64 rows x 4 16B-chunks = 256 chunks (1/thread).
  auto issue_tile = [&](int k0, int buf) {
    for (int q = tid; q < 512; q += 256) {
      const int r = q >> 2, c = q & 3;
      async_copy_b128(&sA[buf][r * LS + c * 8],
                      A + (size_t)(m0 + r) * D_ + k0 + c * 8);
    }
    {
      const int r = tid >> 2, c = tid & 3;
      async_copy_b128(&sB[buf][r * LS + c * 8],
                      Wt + (size_t)(n0 + r) * D_ + k0 + c * 8);
    }
  };

  v8f c[2][2] = {};

  issue_tile(0, 0);
  const int NT = D_ / BK;  // 32 k-steps
  for (int kt = 0; kt < NT; ++kt) {
    const int buf = kt & 1;
    wait_async0();    // my chunks of tile kt have landed in LDS
    __syncthreads();  // everyone's chunks visible; prev reads of buf^1 done
    if (kt + 1 < NT) issue_tile((kt + 1) * BK, buf ^ 1);

    v16h a0 = load_frag_lds<LS>(sA[buf], wr * 32, 0);
    v16h a1 = load_frag_lds<LS>(sA[buf], wr * 32 + 16, 0);
    v16h b0 = load_frag_lds<LS>(sB[buf], wc * 32, 0);
    v16h b1 = load_frag_lds<LS>(sB[buf], wc * 32 + 16, 0);
    c[0][0] = wmma16(a0, b0, c[0][0]);
    c[0][1] = wmma16(a0, b1, c[0][1]);
    c[1][0] = wmma16(a1, b0, c[1][0]);
    c[1][1] = wmma16(a1, b1, c[1][1]);
  }

  // ---- epilogue ----
  // C/D layout: element (vgpr r, lane): row = r + (lane>=16 ? 8 : 0),
  // col = lane & 15 (ISA 7.12.2).
  const int half8 = (lane >> 4) << 3;
  const int ncol = lane & 15;
  for (int i = 0; i < 2; ++i)
    for (int j = 0; j < 2; ++j)
      for (int r = 0; r < 8; ++r) {
        const int m = m0 + wr * 32 + i * 16 + r + half8;
        const int n = n0 + wc * 32 + j * 16 + ncol;
        float v = c[i][j][r] + bias[n];
        if (HEADS) {
          v *= scale;  // 1/sqrt(DK) folded into Q
          const int b = m >> 11, t = m & (T_ - 1);
          const int h = n >> 6, d = n & (DK_ - 1);
          ((_Float16*)outp)[(((size_t)b * H_ + h) * T_ + t) * DK_ + d] =
              (_Float16)v;
        } else {
          ((float*)outp)[(size_t)m * D_ + n] = v;
        }
      }
}

// ---------------------------------------------------------------------------
// Kernel 2: flash-style attention with post-softmax tanh-mask.
//   out[q,:] = sum_k exp(s)*tanh(mask) * V[k,:]  /  sum_k exp(s)
// 4 waves per WG; wave w handles 16 query rows; loop over 64-key tiles.
// ---------------------------------------------------------------------------
#define VTS 72  // LDS row stride in halves for V^T / P tiles (64+8; 144B)

__global__ __launch_bounds__(128) void attn_wmma(
    const _Float16* __restrict__ Qh, const _Float16* __restrict__ Kh,
    const _Float16* __restrict__ Vh, const _Float16* __restrict__ Tm,
    _Float16* __restrict__ ctx) {
  __shared__ _Float16 sVt[DK_ * VTS];    // V^T tile: [d][k], shared by WG
  __shared__ _Float16 sP[4 * 16 * VTS];  // per-wave P tile: [q][k]

  const int tid = threadIdx.x, lane = tid & 31, wid = tid >> 5;
  const int bh = blockIdx.y;  // b*16 + h
  const int b = bh >> 4, h = bh & 15;
  const int q0 = blockIdx.x * 64 + wid * 16;
  const size_t base = (size_t)bh * T_ * DK_;

  // Q fragments (scale already folded in at projection time)
  v16h qa[2];
  {
    const _Float16* qb = Qh + base + (size_t)q0 * DK_;
    qa[0] = load_frag_g64(qb, 0);
    qa[1] = load_frag_g64(qb, 32);
  }

  v8f o[4] = {};
  float rmax[8], rsum[8];
  for (int r = 0; r < 8; ++r) { rmax[r] = -1e30f; rsum[r] = 0.0f; }

  const int half8 = (lane >> 4) << 3;
  const int ncol = lane & 15;
  _Float16* pl = sP + wid * 16 * VTS;

  for (int kt = 0; kt < T_ / 64; ++kt) {
    const int k0 = kt * 64;

    __syncthreads();  // previous iteration's sVt reads complete
    {                 // cooperative transpose of V tile: V[k][d] -> sVt[d][k]
      const int kv = tid & 63;
      const int dh = tid >> 6;  // 0 or 1 (d halves)
      const _Float16* vrow = Vh + base + (size_t)(k0 + kv) * DK_ + dh * 32;
      for (int dd = 0; dd < 32; ++dd)
        sVt[(dh * 32 + dd) * VTS + kv] = vrow[dd];
    }
    __syncthreads();

    // ---- S = Q K^T for this wave's 16 rows x 64 keys ----
    v8f s[4] = {};
    for (int j = 0; j < 4; ++j) {
      const _Float16* kb = Kh + base + (size_t)(k0 + j * 16) * DK_;
      v16h kb0 = load_frag_g64(kb, 0);
      v16h kb1 = load_frag_g64(kb, 32);
      s[j] = wmma16(qa[0], kb0, s[j]);
      s[j] = wmma16(qa[1], kb1, s[j]);
    }

    // ---- online softmax row stats (rows live in (vgpr r, lane-half)) ----
    float scl[8];
    for (int r = 0; r < 8; ++r) {
      float m = fmaxf(fmaxf(s[0][r], s[1][r]), fmaxf(s[2][r], s[3][r]));
      m = fmaxf(m, __shfl_xor(m, 1));
      m = fmaxf(m, __shfl_xor(m, 2));
      m = fmaxf(m, __shfl_xor(m, 4));
      m = fmaxf(m, __shfl_xor(m, 8));
      const float mn = fmaxf(rmax[r], m);
      scl[r] = __expf(rmax[r] - mn);
      rmax[r] = mn;
    }

    // ---- P = exp(S-max); denominator unmasked, numerator * tanh(mask) ----
    for (int r = 0; r < 8; ++r) {
      const int qq = q0 + r + half8;
      float ts = 0.0f;
      for (int j = 0; j < 4; ++j) {
        const float e = __expf(s[j][r] - rmax[r]);
        ts += e;
        const int kk = k0 + j * 16 + ncol;
        const float tm = (float)Tm[(size_t)qq * T_ + kk];
        pl[(r + half8) * VTS + j * 16 + ncol] = (_Float16)(e * tm);
      }
      ts += __shfl_xor(ts, 1);
      ts += __shfl_xor(ts, 2);
      ts += __shfl_xor(ts, 4);
      ts += __shfl_xor(ts, 8);
      rsum[r] = rsum[r] * scl[r] + ts;
      for (int j = 0; j < 4; ++j) o[j][r] *= scl[r];
    }

    // wave-internal LDS RAW: P stores above -> fragment loads below
    asm volatile("s_wait_dscnt 0x0" ::: "memory");

    // ---- O += P @ V ----
    v16h pa0 = load_frag_lds<VTS>(pl, 0, 0);
    v16h pa1 = load_frag_lds<VTS>(pl, 0, 32);
    for (int j = 0; j < 4; ++j) {
      v16h vb0 = load_frag_lds<VTS>(sVt, j * 16, 0);
      v16h vb1 = load_frag_lds<VTS>(sVt, j * 16, 32);
      o[j] = wmma16(pa0, vb0, o[j]);
      o[j] = wmma16(pa1, vb1, o[j]);
    }
  }

  // ---- normalize and write ctx as f16 [B, T, D] (heads re-interleaved) ----
  float inv[8];
  for (int r = 0; r < 8; ++r) inv[r] = 1.0f / fmaxf(rsum[r], 1e-20f);
  for (int j = 0; j < 4; ++j)
    for (int r = 0; r < 8; ++r) {
      const int q = q0 + r + half8;
      const int dcol = h * DK_ + j * 16 + ncol;
      ctx[((size_t)b * T_ + q) * D_ + dcol] = (_Float16)(o[j][r] * inv[r]);
    }
}

// ---------------------------------------------------------------------------
// Host launcher
// ---------------------------------------------------------------------------
extern "C" void kernel_launch(void* const* d_in, const int* in_sizes, int n_in,
                              void* d_out, int out_size, void* d_ws,
                              size_t ws_size, hipStream_t stream) {
  (void)in_sizes; (void)n_in; (void)out_size; (void)ws_size;

  const float* query = (const float*)d_in[0];
  const float* key   = (const float*)d_in[1];
  const float* value = (const float*)d_in[2];
  const float* Wq = (const float*)d_in[3];
  const float* bq = (const float*)d_in[4];
  const float* Wk = (const float*)d_in[5];
  const float* bk = (const float*)d_in[6];
  const float* Wv = (const float*)d_in[7];
  const float* bv = (const float*)d_in[8];
  const float* Wo = (const float*)d_in[9];
  const float* bo = (const float*)d_in[10];
  const float* lmask = (const float*)d_in[11];

  // workspace carve-up (MiB offsets), 64 MiB total
  char* ws = (char*)d_ws;
  const size_t MB = (size_t)1 << 20;
  _Float16* Qh  = (_Float16*)(ws + 0 * MB);   // [B,H,T,64] f16 (pre-scaled)
  _Float16* Kh  = (_Float16*)(ws + 8 * MB);
  _Float16* Vh  = (_Float16*)(ws + 16 * MB);
  _Float16* Tm  = (_Float16*)(ws + 24 * MB);  // tanh(mask) f16 [T,T]
  _Float16* Xq  = (_Float16*)(ws + 32 * MB);  // f16 activations
  _Float16* Xk  = (_Float16*)(ws + 40 * MB);
  _Float16* Xv  = (_Float16*)(ws + 48 * MB);
  _Float16* Wqt = (_Float16*)(ws + 56 * MB);  // f16 transposed weights [n][k]
  _Float16* Wkt = (_Float16*)(ws + 58 * MB);
  _Float16* Wvt = (_Float16*)(ws + 60 * MB);
  _Float16* Wot = (_Float16*)(ws + 62 * MB);
  _Float16* ctx = (_Float16*)(ws + 32 * MB);  // reuse Xq (consumed by then)

  const int nmask = T_ * T_;
  tanh_mask_kernel<<<(nmask + 255) / 256, 256, 0, stream>>>(lmask, Tm, nmask);

  const int n4 = M_ * D_ / 4;  // activation f32->f16, 4 elems/thread
  cvt_f16_kernel<<<(n4 + 255) / 256, 256, 0, stream>>>(query, Xq, n4);
  cvt_f16_kernel<<<(n4 + 255) / 256, 256, 0, stream>>>(key, Xk, n4);
  cvt_f16_kernel<<<(n4 + 255) / 256, 256, 0, stream>>>(value, Xv, n4);

  const dim3 tgrid(D_ / 32, D_ / 32);  // 32 x 32 transpose tiles
  const dim3 tblk(32, 8);
  transpose_cvt_kernel<<<tgrid, tblk, 0, stream>>>(Wq, Wqt);
  transpose_cvt_kernel<<<tgrid, tblk, 0, stream>>>(Wk, Wkt);
  transpose_cvt_kernel<<<tgrid, tblk, 0, stream>>>(Wv, Wvt);
  transpose_cvt_kernel<<<tgrid, tblk, 0, stream>>>(Wo, Wot);

  const dim3 ggrid(M_ / BM, D_ / BN);  // 32 x 16
  const float qscale = 0.125f;         // 1/sqrt(64)
  gemm_wmma_async<true><<<ggrid, 256, 0, stream>>>(Xq, Wqt, bq, Qh, qscale);
  gemm_wmma_async<true><<<ggrid, 256, 0, stream>>>(Xk, Wkt, bk, Kh, 1.0f);
  gemm_wmma_async<true><<<ggrid, 256, 0, stream>>>(Xv, Wvt, bv, Vh, 1.0f);

  attn_wmma<<<dim3(T_ / 64, B_ * H_), 128, 0, stream>>>(Qh, Kh, Vh, Tm, ctx);

  gemm_wmma_async<false><<<ggrid, 256, 0, stream>>>(ctx, Wot, bo, d_out, 1.0f);
}
